// MultiHeadedAttention_38671885533195
// MI455X (gfx1250) — matrix operand order repaired and checked by reading, not compiled, not run on hardware
//
#include <hip/hip_runtime.h>
#include <hip/hip_bf16.h>

typedef __attribute__((ext_vector_type(16))) _Float16 v16h;
typedef __attribute__((ext_vector_type(8)))  _Float16 v8h;
typedef __attribute__((ext_vector_type(8)))  float    v8f;
typedef __attribute__((ext_vector_type(4)))  float    v4f;

#define H_  16
#define DM_ 1024
#define DK_ 64
#define B_  2
#define S_  2048

// ---------------- WMMA fragment loaders (wave32, 16x16x32 f16) ----------------
// A-matrix 16x32 f16 (ISA 7.12.2): lanes 0-15: M=lane, K = kb..kb+7, kb+16..kb+23
// with kb=0; lanes 16-31: same M, kb=8.
__device__ __forceinline__ v16h load_a_f16(const _Float16* A, int ld, int r0, int k0, int lane) {
  int row = r0 + (lane & 15);
  int kb  = k0 + ((lane & 16) ? 8 : 0);
  const _Float16* p = A + (long)row * ld + kb;
  v8h lo = *(const v8h*)(p);
  v8h hi = *(const v8h*)(p + 16);
  v16h a;
#pragma unroll
  for (int i = 0; i < 8; ++i) { a[i] = lo[i]; a[i + 8] = hi[i]; }
  return a;
}

// Non-temporal fp32 A loads for the streaming p_attn reads (P is fp32 by contract).
__device__ __forceinline__ v16h load_a_f32_nt(const float* A, int ld, int r0, int k0, int lane) {
  int row = r0 + (lane & 15);
  int kb  = k0 + ((lane & 16) ? 8 : 0);
  const float* p = A + (long)row * ld + kb;
  v4f x0 = __builtin_nontemporal_load((const v4f*)(p));
  v4f x1 = __builtin_nontemporal_load((const v4f*)(p + 4));
  v4f x2 = __builtin_nontemporal_load((const v4f*)(p + 16));
  v4f x3 = __builtin_nontemporal_load((const v4f*)(p + 20));
  v16h a;
#pragma unroll
  for (int i = 0; i < 4; ++i) {
    a[i]      = (_Float16)x0[i];
    a[i + 4]  = (_Float16)x1[i];
    a[i + 8]  = (_Float16)x2[i];
    a[i + 12] = (_Float16)x3[i];
  }
  return a;
}

// B-matrix 32x16: lane holds column N = lane&15; K = kb..kb+7, kb+16..kb+23
// with kb=0 for lanes 0-15, kb=8 for lanes 16-31 (matches A).  Bt is N x K row-major
// (column-major B).
__device__ __forceinline__ v16h load_b_colmaj_f16(const _Float16* Bt, int ld, int k0, int c0, int lane) {
  int col = c0 + (lane & 15);
  int kb  = k0 + ((lane & 16) ? 8 : 0);
  const _Float16* p = Bt + (long)col * ld + kb;
  v8h lo = *(const v8h*)(p);
  v8h hi = *(const v8h*)(p + 16);
  v16h b;
#pragma unroll
  for (int i = 0; i < 8; ++i) { b[i] = lo[i]; b[i + 8] = hi[i]; }
  return b;
}

__device__ __forceinline__ v8f wmma32(v16h a, v16h b, v8f c) {
  // (neg_a, A, neg_b, B, c_mod, C, reuse_a, reuse_b)
  return __builtin_amdgcn_wmma_f32_16x16x32_f16(false, a, false, b, (short)0, c, false, false);
}

// ---------------- Kernel 0a: activations fp32 -> f16 (streaming, 8 elems/thread) ----------------
__global__ __launch_bounds__(256) void xcvt_kernel(const float* __restrict__ X,
                                                   _Float16* __restrict__ Xh) {
  long i = ((long)blockIdx.x * 256 + threadIdx.x) * 8;
  v4f x0 = *(const v4f*)(X + i);
  v4f x1 = *(const v4f*)(X + i + 4);
  v8h o;
#pragma unroll
  for (int j = 0; j < 4; ++j) { o[j] = (_Float16)x0[j]; o[j + 4] = (_Float16)x1[j]; }
  *(v8h*)(Xh + i) = o;
}

// ---------------- Kernel 0b: weight fp32 [K,N] -> f16 col-major [N,K] ----------------
__global__ __launch_bounds__(256) void wcvt_kernel(const float* __restrict__ W,
                                                   _Float16* __restrict__ Wt) {
  __shared__ float tile[32][33];
  const int bx = blockIdx.x * 32;   // N block
  const int by = blockIdx.y * 32;   // K block
  const int tx = threadIdx.x & 31, ty = threadIdx.x >> 5;   // 32 x 8
#pragma unroll
  for (int i = ty; i < 32; i += 8)
    tile[i][tx] = W[(long)(by + i) * DM_ + bx + tx];        // coalesced read
  __syncthreads();
#pragma unroll
  for (int i = ty; i < 32; i += 8)
    Wt[(long)(bx + i) * DM_ + by + tx] = (_Float16)tile[tx][i];  // coalesced write
}

// ---------------- Kernel 1: projection GEMM, 32x64 per wave ----------------
// Xh: [B*S, DM] f16, Wt: f16 col-major [N,K], out mode 0: f16 [B,H,S,DK],
// mode 1 (for V): f16 [B,H,DK,S] (transposed for col-major B use in P@V).
__global__ __launch_bounds__(128) void proj_kernel(const _Float16* __restrict__ Xh,
                                                   const _Float16* __restrict__ Wt,
                                                   const float* __restrict__ bias,
                                                   _Float16* __restrict__ out,
                                                   int vmode) {
  const int lane = threadIdx.x & 31;
  const int gid  = blockIdx.x * 4 + (threadIdx.x >> 5);
  const int tN = gid & 15;          // 16 col-tiles of 64
  const int tM = gid >> 4;          // 128 row-tiles of 32
  const int r0 = tM * 32, c0 = tN * 64;
  v8f acc[2][4] = {};
  for (int k0 = 0; k0 < DM_; k0 += 32) {
    v16h a0 = load_a_f16(Xh, DM_, r0,      k0, lane);
    v16h a1 = load_a_f16(Xh, DM_, r0 + 16, k0, lane);
#pragma unroll
    for (int j = 0; j < 4; ++j) {
      v16h b = load_b_colmaj_f16(Wt, DM_, k0, c0 + 16 * j, lane);
      acc[0][j] = wmma32(a0, b, acc[0][j]);
      acc[1][j] = wmma32(a1, b, acc[1][j]);
    }
  }
  const int coll = lane & 15;
#pragma unroll
  for (int mi = 0; mi < 2; ++mi) {
    const int rbase = r0 + 16 * mi + ((lane & 16) ? 8 : 0);
#pragma unroll
    for (int j = 0; j < 4; ++j) {
      int colg = c0 + 16 * j + coll;
      int h = colg >> 6, dk = colg & 63;
      float bv = bias[colg];
#pragma unroll
      for (int v = 0; v < 8; ++v) {
        int r = rbase + v;
        int bidx = r >> 11, s = r & (S_ - 1);
        _Float16 val = (_Float16)(acc[mi][j][v] + bv);
        if (vmode)
          out[(((long)bidx * H_ + h) * DK_ + dk) * S_ + s] = val;   // [B,H,DK,S]
        else
          out[(((long)bidx * H_ + h) * S_ + s) * DK_ + dk] = val;   // [B,H,S,DK]
      }
    }
  }
}

// ---------------- Kernel 2: scores = Q @ K^T / 8 -> fp32 p_attn (NT stores) ----------------
__global__ __launch_bounds__(128) void scores_kernel(const _Float16* __restrict__ qh,
                                                     const _Float16* __restrict__ kh,
                                                     float* __restrict__ P) {
  const int lane = threadIdx.x & 31;
  const int gid  = blockIdx.x * 4 + (threadIdx.x >> 5);
  const int bh  = gid >> 11;            // 2048 wave-tiles per (b,h)
  const int rem = gid & 2047;
  const int tN = rem & 31;              // 32 col-tiles of 64
  const int tM = rem >> 5;              // 64 row-tiles of 32
  const int r0 = tM * 32, c0 = tN * 64;
  const _Float16* Q = qh + (long)bh * S_ * DK_;
  const _Float16* K = kh + (long)bh * S_ * DK_;
  v8f acc[2][4] = {};
#pragma unroll
  for (int k0 = 0; k0 < DK_; k0 += 32) {
    v16h a0 = load_a_f16(Q, DK_, r0,      k0, lane);
    v16h a1 = load_a_f16(Q, DK_, r0 + 16, k0, lane);
#pragma unroll
    for (int j = 0; j < 4; ++j) {
      v16h b = load_b_colmaj_f16(K, DK_, k0, c0 + 16 * j, lane);   // K^T
      acc[0][j] = wmma32(a0, b, acc[0][j]);
      acc[1][j] = wmma32(a1, b, acc[1][j]);
    }
  }
  float* Pb = P + (long)bh * S_ * S_;
  const int coll = lane & 15;
#pragma unroll
  for (int mi = 0; mi < 2; ++mi) {
    const int rbase = r0 + 16 * mi + ((lane & 16) ? 8 : 0);
#pragma unroll
    for (int j = 0; j < 4; ++j) {
      int colg = c0 + 16 * j + coll;
#pragma unroll
      for (int v = 0; v < 8; ++v)
        __builtin_nontemporal_store(acc[mi][j][v] * 0.125f,
                                    &Pb[(long)(rbase + v) * S_ + colg]);
    }
  }
}

// ---------------- Kernel 3: masked softmax in place over last dim ----------------
__global__ __launch_bounds__(256) void softmax_kernel(float* __restrict__ P,
                                                      const int* __restrict__ mask) {
  const int tid = threadIdx.x;
  const long row = blockIdx.x;                 // [0, B*H*S)
  const int q  = (int)(row & (S_ - 1));
  const int bh = (int)(row >> 11);
  const int b  = bh >> 4;
  float* p = P + row * S_;
  const int* m = mask + ((long)b * S_ + q) * S_;   // mask[b,0,q,:]
  __shared__ float red[256];
  float mx = -3.0e38f;
  for (int k = tid; k < S_; k += 256) {
    float v = (m[k] == 0) ? -3.0e38f : p[k];
    mx = fmaxf(mx, v);
  }
  red[tid] = mx; __syncthreads();
  for (int s = 128; s > 0; s >>= 1) {
    if (tid < s) red[tid] = fmaxf(red[tid], red[tid + s]);
    __syncthreads();
  }
  mx = red[0];
  __syncthreads();
  float sum = 0.f;
  for (int k = tid; k < S_; k += 256) {
    float e = (m[k] == 0) ? 0.f : __expf(p[k] - mx);
    p[k] = e;
    sum += e;
  }
  red[tid] = sum; __syncthreads();
  for (int s = 128; s > 0; s >>= 1) {
    if (tid < s) red[tid] += red[tid + s];
    __syncthreads();
  }
  sum = red[0];
  float inv = (sum > 0.f) ? 1.f / sum : 0.f;
  for (int k = tid; k < S_; k += 256) p[k] *= inv;
}

// ---------------- Kernel 4: x = p_attn @ V -> f16 [B,S,DM] (NT p reads) ----------------
// Vt: f16 [B,H,DK,S] (col-major B for the GEMM)
__global__ __launch_bounds__(128) void attnv_kernel(const float* __restrict__ P,
                                                    const _Float16* __restrict__ vth,
                                                    _Float16* __restrict__ x) {
  const int lane = threadIdx.x & 31;
  const int gid  = blockIdx.x * 4 + (threadIdx.x >> 5);
  const int bh = gid >> 6;      // 64 row-tiles of 32 per (b,h); N=64 = one tile
  const int tM = gid & 63;
  const int r0 = tM * 32;
  const float* Pb = P + (long)bh * S_ * S_;
  const _Float16* Vt = vth + (long)bh * DK_ * S_;
  v8f acc[2][4] = {};
  for (int k0 = 0; k0 < S_; k0 += 32) {
    v16h a0 = load_a_f32_nt(Pb, S_, r0,      k0, lane);
    v16h a1 = load_a_f32_nt(Pb, S_, r0 + 16, k0, lane);
#pragma unroll
    for (int j = 0; j < 4; ++j) {
      v16h b = load_b_colmaj_f16(Vt, S_, k0, 16 * j, lane);
      acc[0][j] = wmma32(a0, b, acc[0][j]);
      acc[1][j] = wmma32(a1, b, acc[1][j]);
    }
  }
  const int bidx = bh >> 4, h = bh & 15;
  const int coll = lane & 15;
#pragma unroll
  for (int mi = 0; mi < 2; ++mi) {
    const int rbase = r0 + 16 * mi + ((lane & 16) ? 8 : 0);
#pragma unroll
    for (int j = 0; j < 4; ++j) {
      int dk = 16 * j + coll;
#pragma unroll
      for (int v = 0; v < 8; ++v) {
        int s = rbase + v;
        x[((long)bidx * S_ + s) * DM_ + h * DK_ + dk] = (_Float16)acc[mi][j][v];
      }
    }
  }
}

// ---------------- Kernel 5: out = x @ wo + bo -> fp32 d_out ----------------
__global__ __launch_bounds__(128) void outproj_kernel(const _Float16* __restrict__ X,
                                                      const _Float16* __restrict__ Wt,
                                                      const float* __restrict__ bias,
                                                      float* __restrict__ out) {
  const int lane = threadIdx.x & 31;
  const int gid  = blockIdx.x * 4 + (threadIdx.x >> 5);
  const int tN = gid & 15, tM = gid >> 4;
  const int r0 = tM * 32, c0 = tN * 64;
  v8f acc[2][4] = {};
  for (int k0 = 0; k0 < DM_; k0 += 32) {
    v16h a0 = load_a_f16(X, DM_, r0,      k0, lane);
    v16h a1 = load_a_f16(X, DM_, r0 + 16, k0, lane);
#pragma unroll
    for (int j = 0; j < 4; ++j) {
      v16h b = load_b_colmaj_f16(Wt, DM_, k0, c0 + 16 * j, lane);
      acc[0][j] = wmma32(a0, b, acc[0][j]);
      acc[1][j] = wmma32(a1, b, acc[1][j]);
    }
  }
  const int coll = lane & 15;
#pragma unroll
  for (int mi = 0; mi < 2; ++mi) {
    const int rbase = r0 + 16 * mi + ((lane & 16) ? 8 : 0);
#pragma unroll
    for (int j = 0; j < 4; ++j) {
      int colg = c0 + 16 * j + coll;
      float bv = bias[colg];
#pragma unroll
      for (int v = 0; v < 8; ++v)
        out[(long)(rbase + v) * DM_ + colg] = acc[mi][j][v] + bv;
    }
  }
}

extern "C" void kernel_launch(void* const* d_in, const int* in_sizes, int n_in,
                              void* d_out, int out_size, void* d_ws, size_t ws_size,
                              hipStream_t stream) {
  const float* query = (const float*)d_in[0];
  const float* key_  = (const float*)d_in[1];
  const float* value = (const float*)d_in[2];
  const int*   mask  = (const int*)d_in[3];
  const float* wq = (const float*)d_in[4];
  const float* bq = (const float*)d_in[5];
  const float* wk = (const float*)d_in[6];
  const float* bk = (const float*)d_in[7];
  const float* wv = (const float*)d_in[8];
  const float* bv = (const float*)d_in[9];
  const float* wo = (const float*)d_in[10];
  const float* bo = (const float*)d_in[11];

  float* out = (float*)d_out;
  float* P   = out + (long)B_ * S_ * DM_;   // p_attn region of d_out

  // workspace (f16): Q,K [B,H,S,DK]; Vt [B,H,DK,S]; x [B*S,DM];
  // converted activations q/k/v [B*S,DM]; 4 weights col-major [N,K] -> 64 MB
  const long ACT = (long)B_ * S_ * DM_;   // 4 Mi elements
  _Float16* qh  = (_Float16*)d_ws;
  _Float16* kh  = qh  + ACT;
  _Float16* vth = kh  + ACT;
  _Float16* xh  = vth + ACT;
  _Float16* qx  = xh  + ACT;
  _Float16* kx  = qx  + ACT;
  _Float16* vx  = kx  + ACT;
  _Float16* wqt = vx  + ACT;
  _Float16* wkt = wqt + (long)DM_ * DM_;
  _Float16* wvt = wkt + (long)DM_ * DM_;
  _Float16* wot = wvt + (long)DM_ * DM_;

  // activation + weight conversion
  const int cblocks = (int)(ACT / (256 * 8));
  xcvt_kernel<<<cblocks, 256, 0, stream>>>(query, qx);
  xcvt_kernel<<<cblocks, 256, 0, stream>>>(key_,  kx);
  xcvt_kernel<<<cblocks, 256, 0, stream>>>(value, vx);
  dim3 tgrid(DM_ / 32, DM_ / 32);
  wcvt_kernel<<<tgrid, 256, 0, stream>>>(wq, wqt);
  wcvt_kernel<<<tgrid, 256, 0, stream>>>(wk, wkt);
  wcvt_kernel<<<tgrid, 256, 0, stream>>>(wv, wvt);
  wcvt_kernel<<<tgrid, 256, 0, stream>>>(wo, wot);

  dim3 blk(128);
  proj_kernel<<<512, blk, 0, stream>>>(qx, wqt, bq, qh, 0);
  proj_kernel<<<512, blk, 0, stream>>>(kx, wkt, bk, kh, 0);
  proj_kernel<<<512, blk, 0, stream>>>(vx, wvt, bv, vth, 1);
  scores_kernel<<<16384, blk, 0, stream>>>(qh, kh, P);
  softmax_kernel<<<B_ * H_ * S_, 256, 0, stream>>>(P, mask);
  attnv_kernel<<<512, blk, 0, stream>>>(P, vth, xh);
  outproj_kernel<<<512, blk, 0, stream>>>(xh, wot, bo, out);
}
